// RoutingLoRASoftMoe_74715251081275
// MI455X (gfx1250) — matrix operand order repaired and checked by reading, not compiled, not run on hardware
//
#include <hip/hip_runtime.h>

typedef _Float16 half_t;
typedef __attribute__((ext_vector_type(16))) _Float16 v16h;
typedef __attribute__((ext_vector_type(8)))  _Float16 v8h;
typedef __attribute__((ext_vector_type(8)))  float    v8f;
typedef unsigned int u32x4 __attribute__((ext_vector_type(4)));
typedef int          i32x8 __attribute__((ext_vector_type(8)));
typedef int          i32x4 __attribute__((ext_vector_type(4)));

// ---- problem dims (match reference) ----
#define NB    2
#define SEQ   512
#define DIM   1024
#define RNK   16
#define NSK   512            // n_skills == SEQ
#define KRD   (NSK*RNK)      // 8192
#define BSM   (NB*SEQ)       // 1024 token rows

// ---- GEMM tiling: 256 threads = 8 wave32; block tile 64x128, wave 32x32 ----
#define BM 64
#define BN 128
#define BK 32
#define LDST 40              // padded LDS row stride in halves (80B rows; TDM pad reproduces this)

#if defined(__has_builtin)
#  if __has_builtin(__builtin_amdgcn_tensor_load_to_lds) && __has_builtin(__builtin_amdgcn_s_wait_tensorcnt)
#    define HAVE_TDM 1
#  endif
#endif
#ifndef HAVE_TDM
#  define HAVE_TDM 0
#endif

__device__ __forceinline__ int imax(int a, int b) { return a > b ? a : b; }

__device__ __forceinline__ unsigned lds_off(const void* p) {
  // flat shared address: low 32 bits are the wave-relative LDS byte offset
  return (unsigned)(unsigned long long)(uintptr_t)p;
}

__device__ __forceinline__ v16h frag_join(const half_t* lo8, const half_t* hi8) {
  v8h lo = *(const v8h*)lo8;
  v8h hi = *(const v8h*)hi8;
  return __builtin_shufflevector(lo, hi, 0,1,2,3,4,5,6,7,8,9,10,11,12,13,14,15);
}

// A fragment (16x32 f16, MxK). ISA: lane -> row lane&15, h=lane>>4,
// halves j0..7 -> K=8h+j ; j8..15 -> K=16+8h+(j-8). LDS: sA[row][k].
__device__ __forceinline__ v16h load_fragA(const half_t* sA, int rowBase, int lane) {
  const int row = rowBase + (lane & 15);
  const int h   = lane >> 4;
  const half_t* p = sA + row * LDST;
  return frag_join(p + h * 8, p + 16 + h * 8);
}

// B fragment (32x16 f16, KxN). ISA: lane -> N=lane&15, h=lane>>4,
// halves j0..7 -> K=8h+j ; j8..15 -> K=16+8h+(j-8).
// LDS stored transposed: sB[n][k] so the 16 K-halves are contiguous.
__device__ __forceinline__ v16h load_fragB(const half_t* sB, int colBase, int lane) {
  const int n = colBase + (lane & 15);
  const int h = lane >> 4;
  const half_t* p = sB + n * LDST;
  return frag_join(p + h * 8, p + 16 + h * 8);
}

#if HAVE_TDM
// Loop-invariant D# group1: data_size=2B, pad_enable, pad_interval=16 DW (32 halves),
// pad_amount=4 DW (8 halves)  ->  LDS row stride = 40 halves = LDST.
__device__ __forceinline__ i32x8 tdm_g1(int rowLen, int rowsTot, int tileRows) {
  i32x8 g1;
  g1[0] = (1 << 16) | (1 << 20) | (3 << 22) | (3 << 25);
  g1[1] = (int)((unsigned)rowLen << 16);                             // tensor_dim0[15:0]
  g1[2] = (int)(((unsigned)rowLen >> 16) & 0xFFFFu) |
          (int)(((unsigned)rowsTot & 0xFFFFu) << 16);                // td0 hi | td1 lo
  g1[3] = (int)(((unsigned)rowsTot >> 16) & 0xFFFFu) | (32 << 16);   // td1 hi | tile_dim0=32
  g1[4] = (int)(unsigned)tileRows;                                   // tile_dim1 | tile_dim2=0
  g1[5] = rowLen;                                                    // tensor_dim0_stride[31:0]
  g1[6] = 0;
  g1[7] = 0;
  return g1;
}

// Per-step issue: only lds_addr + global_addr vary.
__device__ __forceinline__ void tdm_issue(unsigned ldsOff, unsigned long long ga,
                                          const i32x8& g1) {
  u32x4 g0;
  g0[0] = 1u;                                                  // count=1 (valid user D#)
  g0[1] = ldsOff;                                              // lds_addr
  g0[2] = (unsigned)ga;                                        // global_addr[31:0]
  g0[3] = (unsigned)((ga >> 32) & 0x1FFFFFFull) | (2u << 30);  // global_addr[56:32] | type=2
  const i32x4 z4 = {0, 0, 0, 0};
  const i32x8 z8 = {0, 0, 0, 0, 0, 0, 0, 0};
  __builtin_amdgcn_tensor_load_to_lds(g0, g1, z4, z4, z8, 0);
}
#endif

enum GemmMode { M_MIXING = 0, M_MIXED = 1, M_LORA = 2, M_OUT = 3 };

// All operands are pre-packed f16, A as [m][k], B as [n][k], K-contiguous.
//  M_MIXING: mixing(f32) = xh @ phiT'        (K=DIM)
//  M_MIXED : mixedH(f16) = DwH[b] @ xTh[b]'  (K=SEQ, grid.z=batch)
//  M_LORA  : G(f16)      = (mixedH @ AmatT') * Cm[t, n>>4]   (K=DIM)
//  M_OUT   : out(f32)    = G @ lbT' + xh @ Wh' + bias        (K=KRD then DIM)
template<int MODE>
__global__ __launch_bounds__(256)
void gemm_wmma_k(const void* __restrict__ p0, const void* __restrict__ p1,
                 const void* __restrict__ p2, const void* __restrict__ p3,
                 const void* __restrict__ p4, void* __restrict__ outp)
{
  (void)p2; (void)p3; (void)p4;
  __shared__ __align__(16) half_t sA[2][BM * LDST];
  __shared__ __align__(16) half_t sB[2][BN * LDST];

  const int tid   = threadIdx.x;
  const int lane  = tid & 31;
  const int wid   = tid >> 5;
  const int wM    = (wid & 1) * 32;
  const int wN    = (wid >> 1) * 32;
  const int tileM = blockIdx.x * BM;
  const int tileN = blockIdx.y * BN;
  const int bz    = blockIdx.z;
  (void)bz;

  v8f acc[2][2] = {};

  const int nPass = (MODE == M_OUT) ? 2 : 1;
  for (int pass = 0; pass < nPass; ++pass) {
    const half_t* Ag; const half_t* Bg; int lda, ldb, Kd, aRows, bRows;
    if constexpr (MODE == M_MIXING) {
      Ag = (const half_t*)p0;            lda = DIM; aRows = BSM;
      Bg = (const half_t*)p1;            ldb = DIM; bRows = NSK;  Kd = DIM;
    } else if constexpr (MODE == M_MIXED) {
      Ag = (const half_t*)p0 + bz*SEQ*SEQ; lda = SEQ; aRows = SEQ;
      Bg = (const half_t*)p1 + bz*DIM*SEQ; ldb = SEQ; bRows = DIM; Kd = SEQ;
    } else if constexpr (MODE == M_LORA) {
      Ag = (const half_t*)p0;            lda = DIM; aRows = BSM;
      Bg = (const half_t*)p1;            ldb = DIM; bRows = KRD;  Kd = DIM;
    } else {
      if (pass == 0) { Ag = (const half_t*)p0; lda = KRD; aRows = BSM;
                       Bg = (const half_t*)p1; ldb = KRD; bRows = DIM; Kd = KRD; }
      else           { Ag = (const half_t*)p2; lda = DIM; aRows = BSM;
                       Bg = (const half_t*)p3; ldb = DIM; bRows = DIM; Kd = DIM; }
    }
    const int nSteps = Kd / BK;

#if HAVE_TDM
    // ---- TDM double-buffered pipeline: wave0 drives the DMA.
    // Descriptors hoisted: g1 (dims/strides/pad) is loop-invariant; per step only
    // the 64-byte-advancing global address and the ping-pong LDS offset change.
    const unsigned offA0 = lds_off(&sA[0][0]), offA1 = lds_off(&sA[1][0]);
    const unsigned offB0 = lds_off(&sB[0][0]), offB1 = lds_off(&sB[1][0]);
    const i32x8 g1A = tdm_g1(lda, aRows, BM);
    const i32x8 g1B = tdm_g1(ldb, bRows, BN);
    const unsigned long long gaA =
        (unsigned long long)(uintptr_t)Ag + 2ull * (unsigned long long)tileM * (unsigned)lda;
    const unsigned long long gaB =
        (unsigned long long)(uintptr_t)Bg + 2ull * (unsigned long long)tileN * (unsigned)ldb;
    if (tid < 32) {
      tdm_issue(offA0, gaA, g1A);
      tdm_issue(offB0, gaB, g1B);
    }
    for (int s = 0; s < nSteps; ++s) {
      const int  cur  = s & 1;
      const bool more = (s + 1) < nSteps;
      if (tid < 32) {
        if (more) {
          const unsigned long long kb = 64ull * (unsigned long long)(s + 1);  // BK halves = 64 B
          tdm_issue(cur ? offA0 : offA1, gaA + kb, g1A);
          tdm_issue(cur ? offB0 : offB1, gaB + kb, g1B);
          __builtin_amdgcn_s_wait_tensorcnt(2);   // current pair complete (in-order)
        } else {
          __builtin_amdgcn_s_wait_tensorcnt(0);
        }
      }
      __syncthreads();
      const half_t* cA = sA[cur]; const half_t* cB = sB[cur];
      const v16h af0 = load_fragA(cA, wM,      lane);
      const v16h af1 = load_fragA(cA, wM + 16, lane);
      const v16h bf0 = load_fragB(cB, wN,      lane);
      const v16h bf1 = load_fragB(cB, wN + 16, lane);
      acc[0][0] = __builtin_amdgcn_wmma_f32_16x16x32_f16(false, af0, false, bf0, (short)0, acc[0][0], false, false);
      acc[0][1] = __builtin_amdgcn_wmma_f32_16x16x32_f16(false, af0, false, bf1, (short)0, acc[0][1], false, false);
      acc[1][0] = __builtin_amdgcn_wmma_f32_16x16x32_f16(false, af1, false, bf0, (short)0, acc[1][0], false, false);
      acc[1][1] = __builtin_amdgcn_wmma_f32_16x16x32_f16(false, af1, false, bf1, (short)0, acc[1][1], false, false);
      __syncthreads();   // everyone done reading 'cur' before it is re-filled
    }
#else
    // ---- fallback: cooperative vectorized copy (everything is f16 [row][k]) ----
    for (int s = 0; s < nSteps; ++s) {
      const int k0 = s * BK;
      { // A: 64 rows x 4 v8h chunks = 256 chunks
        const int r = tid >> 2, c8 = (tid & 3) * 8;
        *(v8h*)&sA[0][r * LDST + c8] = *(const v8h*)(Ag + (tileM + r) * lda + k0 + c8);
      }
      { // B: 128 rows x 4 chunks = 512 chunks
        for (int e = tid; e < 512; e += 256) {
          const int n = e >> 2, c8 = (e & 3) * 8;
          *(v8h*)&sB[0][n * LDST + c8] = *(const v8h*)(Bg + (tileN + n) * ldb + k0 + c8);
        }
      }
      __syncthreads();
      const v16h af0 = load_fragA(sA[0], wM,      lane);
      const v16h af1 = load_fragA(sA[0], wM + 16, lane);
      const v16h bf0 = load_fragB(sB[0], wN,      lane);
      const v16h bf1 = load_fragB(sB[0], wN + 16, lane);
      acc[0][0] = __builtin_amdgcn_wmma_f32_16x16x32_f16(false, af0, false, bf0, (short)0, acc[0][0], false, false);
      acc[0][1] = __builtin_amdgcn_wmma_f32_16x16x32_f16(false, af0, false, bf1, (short)0, acc[0][1], false, false);
      acc[1][0] = __builtin_amdgcn_wmma_f32_16x16x32_f16(false, af1, false, bf0, (short)0, acc[1][0], false, false);
      acc[1][1] = __builtin_amdgcn_wmma_f32_16x16x32_f16(false, af1, false, bf1, (short)0, acc[1][1], false, false);
      __syncthreads();
    }
#endif
  }

  // ---- epilogue. C/D f32 layout: lane -> N = lane&15, VGPR i -> M = i + 8*(lane>>4)
  const int mOff = 8 * (lane >> 4);
  const int nOff = lane & 15;
  for (int mi = 0; mi < 2; ++mi) {
    for (int ni = 0; ni < 2; ++ni) {
      const int baseM = tileM + wM + mi * 16 + mOff;
      const int gn    = tileN + wN + ni * 16 + nOff;
#pragma unroll
      for (int i = 0; i < 8; ++i) {
        const int gm = baseM + i;
        const float v = acc[mi][ni][i];
        if constexpr (MODE == M_MIXING) {
          ((float*)outp)[gm * NSK + gn] = v;
        } else if constexpr (MODE == M_MIXED) {
          ((half_t*)outp)[(bz * SEQ + gm) * DIM + gn] = (half_t)v;
        } else if constexpr (MODE == M_LORA) {
          const float cs = ((const float*)p2)[gm * NSK + (gn >> 4)];   // Cm[t, k]
          ((half_t*)outp)[gm * KRD + gn] = (half_t)(v * cs);
        } else {
          ((float*)outp)[gm * DIM + gn] = v + ((const float*)p4)[gn];  // + bias
        }
      }
    }
  }
}

// ---------------- packing kernels (one-time f16 repack / transposes) ----------------
__global__ __launch_bounds__(256) void pack_cvt_k(const float* __restrict__ in,
                                                  half_t* __restrict__ out, int n) {
  for (int i = blockIdx.x * 256 + threadIdx.x; i < n; i += gridDim.x * 256)
    out[i] = (half_t)in[i];
}
__global__ __launch_bounds__(256) void pack_phiT_k(const float* __restrict__ phi,
                                                   half_t* __restrict__ o) {
  for (int i = blockIdx.x * 256 + threadIdx.x; i < NSK * DIM; i += gridDim.x * 256) {
    const int n = i / DIM, k = i % DIM;
    o[i] = (half_t)phi[k * NSK + n];
  }
}
__global__ __launch_bounds__(256) void pack_xT_k(const float* __restrict__ x,
                                                 half_t* __restrict__ o) {
  for (int i = blockIdx.x * 256 + threadIdx.x; i < NB * DIM * SEQ; i += gridDim.x * 256) {
    const int b = i / (DIM * SEQ), r = i % (DIM * SEQ);
    const int d = r / SEQ, s = r % SEQ;
    o[i] = (half_t)x[(b * SEQ + s) * DIM + d];
  }
}
__global__ __launch_bounds__(256) void pack_amatT_k(const float* __restrict__ la,
                                                    half_t* __restrict__ o) {
  for (int i = blockIdx.x * 256 + threadIdx.x; i < KRD * DIM; i += gridDim.x * 256) {
    const int c = i / DIM, d = i % DIM;     // AmatT[c][d] = lora_a[c>>4][d][c&15]
    o[i] = (half_t)la[(c >> 4) * (DIM * RNK) + d * RNK + (c & 15)];
  }
}
__global__ __launch_bounds__(256) void pack_lbT_k(const float* __restrict__ lb,
                                                  half_t* __restrict__ o) {
  for (int i = blockIdx.x * 256 + threadIdx.x; i < DIM * KRD; i += gridDim.x * 256) {
    const int d = i / KRD, kr = i % KRD;    // lbT[d][kr] = lora_b[kr][d]
    o[i] = (half_t)lb[kr * DIM + d];
  }
}

// ---------------- mask + dispatch/combine softmax ----------------
// first = argmax(inst), n = sum(inst), last = first + n
// m[p]  = (last + max(p+1-last, 0)) * pad[b,p];  allowed[p,s] = s < m[p]*pad[b,s]
__global__ __launch_bounds__(256)
void router_mask_softmax_k(const float* __restrict__ mixing,
                           const int* __restrict__ inst,
                           const int* __restrict__ pad,
                           half_t* __restrict__ DwH,
                           float* __restrict__ Cm)
{
  __shared__ float srow[SEQ];
  __shared__ float sred[256];
  __shared__ int   ired[256];
  const int p = blockIdx.x, b = blockIdx.y, tid = threadIdx.x;

  int ones = 0, firstIdx = 0x7fffffff;
  for (int j = tid; j < SEQ; j += 256) {
    const int v = inst[b * SEQ + j];
    ones += v;
    if (v > 0 && j < firstIdx) firstIdx = j;
  }
  ired[tid] = ones; __syncthreads();
  for (int s = 128; s > 0; s >>= 1) { if (tid < s) ired[tid] += ired[tid + s]; __syncthreads(); }
  const int nOnes = ired[0]; __syncthreads();
  ired[tid] = firstIdx; __syncthreads();
  for (int s = 128; s > 0; s >>= 1) { if (tid < s) ired[tid] = (ired[tid] < ired[tid + s]) ? ired[tid] : ired[tid + s]; __syncthreads(); }
  int first = ired[0]; __syncthreads();
  if (first == 0x7fffffff) first = 0;

  const int last = first + nOnes;
  const int padP = pad[b * SEQ + p];
  const int mp   = (last + imax(p + 1 - last, 0)) * padP;
  const float padf = (float)padP;

  const float* mrow = mixing + (size_t)(b * SEQ + p) * NSK;
  for (int j = tid; j < SEQ; j += 256) srow[j] = mrow[j];
  __syncthreads();

  // masked dispatch softmax -> DwH (f16)
  float mx = -3.0e38f;
  for (int j = tid; j < SEQ; j += 256) {
    const bool allowed = j < mp * pad[b * SEQ + j];
    mx = fmaxf(mx, allowed ? srow[j] : -3.0e38f);
  }
  sred[tid] = mx; __syncthreads();
  for (int s = 128; s > 0; s >>= 1) { if (tid < s) sred[tid] = fmaxf(sred[tid], sred[tid + s]); __syncthreads(); }
  mx = sred[0]; __syncthreads();
  float sum = 0.f;
  for (int j = tid; j < SEQ; j += 256) {
    const bool allowed = j < mp * pad[b * SEQ + j];
    sum += __expf((allowed ? srow[j] : -3.0e38f) - mx);
  }
  sred[tid] = sum; __syncthreads();
  for (int s = 128; s > 0; s >>= 1) { if (tid < s) sred[tid] += sred[tid + s]; __syncthreads(); }
  sum = sred[0]; __syncthreads();
  {
    const float inv = padf / sum;
    half_t* drow = DwH + (size_t)(b * SEQ + p) * SEQ;
    for (int j = tid; j < SEQ; j += 256) {
      const bool allowed = j < mp * pad[b * SEQ + j];
      drow[j] = (half_t)(__expf((allowed ? srow[j] : -3.0e38f) - mx) * inv);
    }
  }
  __syncthreads();

  // unmasked combine softmax -> Cm (f32)
  float mx2 = -3.0e38f;
  for (int j = tid; j < SEQ; j += 256) mx2 = fmaxf(mx2, srow[j]);
  sred[tid] = mx2; __syncthreads();
  for (int s = 128; s > 0; s >>= 1) { if (tid < s) sred[tid] = fmaxf(sred[tid], sred[tid + s]); __syncthreads(); }
  mx2 = sred[0]; __syncthreads();
  float sum2 = 0.f;
  for (int j = tid; j < SEQ; j += 256) sum2 += __expf(srow[j] - mx2);
  sred[tid] = sum2; __syncthreads();
  for (int s = 128; s > 0; s >>= 1) { if (tid < s) sred[tid] += sred[tid + s]; __syncthreads(); }
  sum2 = sred[0]; __syncthreads();
  {
    const float inv2 = 1.0f / sum2;
    float* crow = Cm + (size_t)(b * SEQ + p) * NSK;
    for (int j = tid; j < SEQ; j += 256) crow[j] = __expf(srow[j] - mx2) * inv2;
  }
}

extern "C" void kernel_launch(void* const* d_in, const int* in_sizes, int n_in,
                              void* d_out, int out_size, void* d_ws, size_t ws_size,
                              hipStream_t stream) {
  (void)in_sizes; (void)n_in; (void)out_size; (void)ws_size;
  const float* x    = (const float*)d_in[0];
  const int*   inst = (const int*)d_in[1];
  const int*   pad  = (const int*)d_in[2];
  const float* phi  = (const float*)d_in[3];
  const float* la   = (const float*)d_in[4];
  const float* lb   = (const float*)d_in[5];
  const float* W    = (const float*)d_in[6];
  const float* bias = (const float*)d_in[7];
  float* out = (float*)d_out;

  char* ws = (char*)d_ws;
  float*  mixing = (float*)(ws + (size_t)( 0ull << 20));  //  2 MB (BSM x NSK) f32
  float*  Cm     = (float*)(ws + (size_t)( 2ull << 20));  //  2 MB (B,S,K)     f32
  half_t* DwH    = (half_t*)(ws + (size_t)( 4ull << 20)); //  1 MB (B,S,S)     f16
  half_t* mixedH = (half_t*)(ws + (size_t)( 5ull << 20)); //  2 MB (BSM x D)   f16
  half_t* G      = (half_t*)(ws + (size_t)( 7ull << 20)); // 16 MB (BSM x KRD) f16
  half_t* xh     = (half_t*)(ws + (size_t)(23ull << 20)); //  2 MB x as [t][d] f16
  half_t* phiT   = (half_t*)(ws + (size_t)(25ull << 20)); //  1 MB phi^T [n][k]
  half_t* xTh    = (half_t*)(ws + (size_t)(26ull << 20)); //  2 MB x^T  [b][d][s]
  half_t* Wh     = (half_t*)(ws + (size_t)(28ull << 20)); //  2 MB W    [d][e]
  half_t* AmatT  = (half_t*)(ws + (size_t)(30ull << 20)); // 16 MB Amat^T [c][d]
  half_t* lbT    = (half_t*)(ws + (size_t)(46ull << 20)); // 16 MB lora_b^T [d][kr]

  const dim3 blk(256);

  // 0) pack all GEMM operands to f16 in [row][K-contiguous] layouts
  pack_cvt_k  <<<dim3(1024), blk, 0, stream>>>(x, xh, BSM * DIM);
  pack_cvt_k  <<<dim3(1024), blk, 0, stream>>>(W, Wh, DIM * DIM);
  pack_phiT_k <<<dim3(512),  blk, 0, stream>>>(phi, phiT);
  pack_xT_k   <<<dim3(1024), blk, 0, stream>>>(x, xTh);
  pack_amatT_k<<<dim3(4096), blk, 0, stream>>>(la, AmatT);
  pack_lbT_k  <<<dim3(4096), blk, 0, stream>>>(lb, lbT);

  // 1) mixing = X @ phi
  gemm_wmma_k<M_MIXING><<<dim3(BSM / BM, NSK / BN, 1), blk, 0, stream>>>(
      xh, phiT, nullptr, nullptr, nullptr, mixing);

  // 2) mask + dispatch/combine softmaxes
  router_mask_softmax_k<<<dim3(SEQ, NB), blk, 0, stream>>>(mixing, inst, pad, DwH, Cm);

  // 3) mixed[b] = Dw[b] @ X[b]
  gemm_wmma_k<M_MIXED><<<dim3(SEQ / BM, DIM / BN, NB), blk, 0, stream>>>(
      DwH, xTh, nullptr, nullptr, nullptr, mixedH);

  // 4) G = (mixed @ Amat) * C[t, col>>4]
  gemm_wmma_k<M_LORA><<<dim3(BSM / BM, KRD / BN, 1), blk, 0, stream>>>(
      mixedH, AmatT, Cm, nullptr, nullptr, G);

  // 5) out = G @ lora_b + X @ W^T + bias
  gemm_wmma_k<M_OUT><<<dim3(BSM / BM, DIM / BN, 1), blk, 0, stream>>>(
      G, lbT, xh, Wh, bias, out);
}
